// HeteroCrossAttention_54211077210529
// MI455X (gfx1250) — matrix operand order repaired and checked by reading, not compiled, not run on hardware
//
#include <hip/hip_runtime.h>

#define DEV static __device__ __forceinline__

typedef __bf16 v16bf __attribute__((ext_vector_type(16)));
typedef __bf16 v8bf  __attribute__((ext_vector_type(8)));
typedef __bf16 v4bf  __attribute__((ext_vector_type(4)));
typedef float  v8f   __attribute__((ext_vector_type(8)));

static constexpr int T  = 2;
static constexpr int NN = 2048;
static constexpr int D  = 256;
static constexpr int H  = 8;
static constexpr int HD = 32;

DEV v16bf make16(v8bf lo, v8bf hi) {
    union { v16bf v; v8bf h[2]; } u;
    u.h[0] = lo; u.h[1] = hi;
    return u.v;
}

DEV v8f wmma_bf16(v16bf a, v16bf b, v8f c) {
    // D(16x16,f32) = A(16x32,bf16) * B(32x16,bf16) + C
    return __builtin_amdgcn_wmma_f32_16x16x32_bf16(false, a, false, b, (short)0, c, false, false);
}

// ---- async global->LDS staging (gfx1250 ASYNCcnt path) ----------------------
// For shared pointers, the generic address low 32 bits are the LDS offset.
DEV unsigned lds_off(const void* p) { return (unsigned)(uintptr_t)p; }

DEV void async_ld_b128(const void* g, unsigned loff) {
    asm volatile("global_load_async_to_lds_b128 %0, %1, off"
                 :: "v"(loff), "v"((unsigned long long)g) : "memory");
}
DEV void async_ld_b64(const void* g, unsigned loff) {
    asm volatile("global_load_async_to_lds_b64 %0, %1, off"
                 :: "v"(loff), "v"((unsigned long long)g) : "memory");
}
DEV void wait_async0() {
    asm volatile("s_wait_asynccnt 0x0" ::: "memory");
}

// A-fragment from a row-major tile with 32 bf16 per row.
// lane L<16 : row, K = {0..7, 16..23}; lane L>=16 : row, K = {8..15, 24..31}
DEV v16bf ld_afrag(const __bf16* base, int row, int lane) {
    const int kH = (lane >> 4) * 8;
    const __bf16* r = base + row * 32;
    return make16(*(const v8bf*)(r + kH), *(const v8bf*)(r + 16 + kH));
}

// B-fragment from a K-transposed tile: tile[col][k], 32 k per row.
// lane L<16 : col, K = {0..7, 16..23} ; lane L>=16 : col, K = {8..15, 24..31}
DEV v16bf ld_bfrag(const __bf16* base, int col, int lane) {
    const int kB = (lane >> 4) * 8;
    const __bf16* r = base + col * 32 + kB;
    return make16(*(const v8bf*)r, *(const v8bf*)(r + 16));
}

DEV v8f zero8() { v8f z = {0.f,0.f,0.f,0.f,0.f,0.f,0.f,0.f}; return z; }

// ---------------------------------------------------------------- convert ---
__global__ void __launch_bounds__(256) convert_kernel(const float* __restrict__ in,
                                                      __bf16* __restrict__ out, int n) {
    int i = blockIdx.x * 256 + threadIdx.x;
    if (i < n) out[i] = (__bf16)in[i];
}

// ---------------------------------------------------------------- GEMM core -
// C(2048x256) = A(2048x256) * W(256x256). Block tile 128x64, K-step 32.
// 8 waves: waveM = wid&3 (4 x 32 rows), waveN = wid>>2 (2 x 32 cols).
DEV void gemm_core(const __bf16* __restrict__ A, const __bf16* __restrict__ W,
                   __bf16* a_lds, __bf16* bt_lds, v8f acc[2][2],
                   int blockM, int blockN) {
    const int tid  = threadIdx.x;
    const int lane = tid & 31;
    const int wid  = tid >> 5;
    const int n16  = lane & 15;
    const int waveM = wid & 3, waveN = wid >> 2;

    for (int kt = 0; kt < 8; ++kt) {
        const int k0 = kt * 32;
        __syncthreads();
        {   // stage A tile 128x32 via async global->LDS (no VGPR bounce)
            const int r = tid >> 1, sgm = (tid & 1) * 16;
            const __bf16* p = A + (size_t)(blockM * 128 + r) * D + k0 + sgm;
            const unsigned la = lds_off(&a_lds[r * 32 + sgm]);
            async_ld_b128(p, la);
            async_ld_b128(p + 8, la + 16);
            if (kt < 7) __builtin_prefetch(p + 32, 0, 0);
        }
        {   // stage W tile 32x64, transposed in-register -> bt_lds[col][k]
            const int wr = tid >> 3, wc = (tid & 7) * 8;
            const __bf16* p = W + (size_t)(k0 + wr) * D + blockN * 64 + wc;
            v8bf w = *(const v8bf*)p;
            if (kt < 7) __builtin_prefetch(p + 32 * D, 0, 0);
#pragma unroll
            for (int j = 0; j < 8; ++j) bt_lds[(wc + j) * 32 + wr] = w[j];
        }
        wait_async0();
        __syncthreads();
#pragma unroll
        for (int sm = 0; sm < 2; ++sm) {
            v16bf af = ld_afrag(a_lds, waveM * 32 + sm * 16 + n16, lane);
#pragma unroll
            for (int sn = 0; sn < 2; ++sn) {
                v16bf bf = ld_bfrag(bt_lds, waveN * 32 + sn * 16 + n16, lane);
                acc[sm][sn] = wmma_bf16(af, bf, acc[sm][sn]);
            }
        }
    }
}

// ------------------------------------------------------------ projections ---
// grid: (16 Mblk, 4 Nblk, T*3). z -> t = z/3, which = z%3 (Q,K,V)
__global__ void __launch_bounds__(256)
proj_kernel(const __bf16* __restrict__ xb,
            const __bf16* __restrict__ wq, const __bf16* __restrict__ wk,
            const __bf16* __restrict__ wv,
            const float* __restrict__ bq, const float* __restrict__ bk,
            const float* __restrict__ bv,
            __bf16* __restrict__ Qb, __bf16* __restrict__ Kb,
            __bf16* __restrict__ Vb) {
    __shared__ __attribute__((aligned(16))) __bf16 a_lds[128 * 32];
    __shared__ __attribute__((aligned(16))) __bf16 bt_lds[64 * 32];

    const int z = blockIdx.z, t = z / 3, which = z % 3;
    const __bf16* W   = (which == 0 ? wq : (which == 1 ? wk : wv)) + (size_t)t * D * D;
    const float*  bia = (which == 0 ? bq : (which == 1 ? bk : bv)) + (size_t)t * D;
    __bf16*       out = (which == 0 ? Qb : (which == 1 ? Kb : Vb)) + (size_t)t * NN * D;
    const __bf16* A   = xb + (size_t)t * NN * D;

    v8f acc[2][2];
#pragma unroll
    for (int i = 0; i < 2; ++i)
#pragma unroll
        for (int j = 0; j < 2; ++j) acc[i][j] = zero8();

    gemm_core(A, W, a_lds, bt_lds, acc, blockIdx.x, blockIdx.y);

    const int lane = threadIdx.x & 31, wid = threadIdx.x >> 5;
    const int n16 = lane & 15, half = lane >> 4;
    const int waveM = wid & 3, waveN = wid >> 2;
#pragma unroll
    for (int sm = 0; sm < 2; ++sm)
#pragma unroll
        for (int sn = 0; sn < 2; ++sn) {
            const int gCol = blockIdx.y * 64 + waveN * 32 + sn * 16 + n16;
            const float b = bia[gCol];
#pragma unroll
            for (int v = 0; v < 8; ++v) {
                const int gRow = blockIdx.x * 128 + waveM * 32 + sm * 16 + half * 8 + v;
                out[(size_t)gRow * D + gCol] = (__bf16)(acc[sm][sn][v] + b);
            }
        }
}

// -------------------------------------------------------------- attention ---
// grid: (16 qblocks of 128 rows, H, T). Each wave: 16 query rows, flash loop
// over 32-key blocks. K/V for head h come from t^1 (the [::-1]).
__global__ void __launch_bounds__(256)
attn_kernel(const __bf16* __restrict__ Qb, const __bf16* __restrict__ Kb,
            const __bf16* __restrict__ Vb, __bf16* __restrict__ attb) {
    __shared__ __attribute__((aligned(16))) __bf16 k_lds[32 * 32];   // [key][d]
    __shared__ __attribute__((aligned(16))) __bf16 vt_lds[32 * 32];  // [d][key]
    __shared__ __attribute__((aligned(16))) __bf16 p_lds[8 * 16 * 32];

    const int tid = threadIdx.x, lane = tid & 31, wid = tid >> 5;
    const int n16 = lane & 15, half = lane >> 4;
    const int t = blockIdx.z, h = blockIdx.y;
    const int kvt = 1 - t;
    const int qBase = blockIdx.x * 128 + wid * 16;
    const float scale = 0.17677669529663687f;  // HD^-0.5

    const size_t qOfs  = (size_t)t   * NN * D;
    const size_t kvOfs = (size_t)kvt * NN * D;

    // Q A-fragment (16 rows x 32 d) — loaded once
    const __bf16* qp = Qb + qOfs + (size_t)(qBase + n16) * D + h * HD + half * 8;
    const v16bf qa = make16(*(const v8bf*)qp, *(const v8bf*)(qp + 16));

    float mrun[8], lrun[8];
#pragma unroll
    for (int v = 0; v < 8; ++v) { mrun[v] = -1e30f; lrun[v] = 0.f; }
    v8f o0 = zero8(), o1 = zero8();

    for (int j0 = 0; j0 < NN; j0 += 32) {
        __syncthreads();
        {   // K tile: async global->LDS (row-major). V tile: transpose via VGPRs.
            const int r = tid >> 3, c = (tid & 7) * 4;
            const __bf16* kp = Kb + kvOfs + (size_t)(j0 + r) * D + h * HD + c;
            async_ld_b64(kp, lds_off(&k_lds[r * 32 + c]));
            const __bf16* vp = Vb + kvOfs + (size_t)(j0 + r) * D + h * HD + c;
            v4bf vv = *(const v4bf*)vp;
            if (j0 + 32 < NN) {
                __builtin_prefetch(kp + 32 * D, 0, 0);
                __builtin_prefetch(vp + 32 * D, 0, 0);
            }
#pragma unroll
            for (int j = 0; j < 4; ++j) vt_lds[(c + j) * 32 + r] = vv[j];
        }
        wait_async0();
        __syncthreads();

        // S = Q * K^T : keys j0+0..15 and j0+16..31
        v16bf kb0 = ld_bfrag(k_lds, n16, lane);
        v16bf kb1 = ld_bfrag(k_lds, 16 + n16, lane);
        v8f s0 = wmma_bf16(qa, kb0, zero8());
        v8f s1 = wmma_bf16(qa, kb1, zero8());

        __bf16* pw = p_lds + wid * (16 * 32);
#pragma unroll
        for (int v = 0; v < 8; ++v) {
            float a0 = s0[v] * scale, a1 = s1[v] * scale;
            float mx = fmaxf(a0, a1);
#pragma unroll
            for (int msk = 1; msk < 16; msk <<= 1) mx = fmaxf(mx, __shfl_xor(mx, msk));
            const float mN   = fmaxf(mrun[v], mx);
            const float corr = __expf(mrun[v] - mN);
            const float p0 = __expf(a0 - mN), p1 = __expf(a1 - mN);
            float ps = p0 + p1;
#pragma unroll
            for (int msk = 1; msk < 16; msk <<= 1) ps += __shfl_xor(ps, msk);
            lrun[v] = lrun[v] * corr + ps;
            mrun[v] = mN;
            o0[v] *= corr; o1[v] *= corr;
            const int row = half * 8 + v;               // D-frag row for this lane/vgpr
            pw[row * 32 + n16]      = (__bf16)p0;
            pw[row * 32 + 16 + n16] = (__bf16)p1;
        }
        // same-wave LDS ops are in-order; just stop the compiler reordering
        asm volatile("" ::: "memory");

        // O += P * V (two 16-wide d slabs)
        v16bf pa  = ld_afrag(pw, n16, lane);
        v16bf vb0 = ld_bfrag(vt_lds, n16, lane);
        v16bf vb1 = ld_bfrag(vt_lds, 16 + n16, lane);
        o0 = wmma_bf16(pa, vb0, o0);
        o1 = wmma_bf16(pa, vb1, o1);
    }

    __bf16* op = attb + qOfs + (size_t)qBase * D + h * HD;
#pragma unroll
    for (int v = 0; v < 8; ++v) {
        const float inv = 1.f / lrun[v];
        const int row = half * 8 + v;
        op[(size_t)row * D + n16]      = (__bf16)(o0[v] * inv);
        op[(size_t)row * D + 16 + n16] = (__bf16)(o1[v] * inv);
    }
}

// -------------------------------------------------- output proj + residual --
__global__ void __launch_bounds__(256)
oproj_kernel(const __bf16* __restrict__ attb, const __bf16* __restrict__ wo,
             const float* __restrict__ bo, const float* __restrict__ x,
             float* __restrict__ y) {
    __shared__ __attribute__((aligned(16))) __bf16 a_lds[128 * 32];
    __shared__ __attribute__((aligned(16))) __bf16 bt_lds[64 * 32];

    const int t = blockIdx.z;
    const __bf16* A = attb + (size_t)t * NN * D;
    const __bf16* W = wo + (size_t)t * D * D;
    const float*  b = bo + (size_t)t * D;
    const float*  xr = x + (size_t)t * NN * D;
    float*        yr = y + (size_t)t * NN * D;

    v8f acc[2][2];
#pragma unroll
    for (int i = 0; i < 2; ++i)
#pragma unroll
        for (int j = 0; j < 2; ++j) acc[i][j] = zero8();

    gemm_core(A, W, a_lds, bt_lds, acc, blockIdx.x, blockIdx.y);

    const int lane = threadIdx.x & 31, wid = threadIdx.x >> 5;
    const int n16 = lane & 15, half = lane >> 4;
    const int waveM = wid & 3, waveN = wid >> 2;
#pragma unroll
    for (int sm = 0; sm < 2; ++sm)
#pragma unroll
        for (int sn = 0; sn < 2; ++sn) {
            const int gCol = blockIdx.y * 64 + waveN * 32 + sn * 16 + n16;
            const float bb = b[gCol];
#pragma unroll
            for (int v = 0; v < 8; ++v) {
                const int gRow = blockIdx.x * 128 + waveM * 32 + sm * 16 + half * 8 + v;
                const size_t o = (size_t)gRow * D + gCol;
                yr[o] = acc[sm][sn][v] + bb + xr[o];
            }
        }
}

// --------------------------------------------------------------- layernorm --
__global__ void __launch_bounds__(256)
ln_kernel(const float* __restrict__ y, const float* __restrict__ g,
          const float* __restrict__ b, float* __restrict__ out) {
    const int lane = threadIdx.x & 31, wid = threadIdx.x >> 5;
    const int row = blockIdx.x * 8 + wid;           // 0..4095
    const int t = row >> 11;                        // row / 2048
    const float* yr = y + (size_t)row * D + lane * 8;

    float4 v0 = *(const float4*)yr;
    float4 v1 = *(const float4*)(yr + 4);
    float s  = v0.x + v0.y + v0.z + v0.w + v1.x + v1.y + v1.z + v1.w;
    float sq = v0.x*v0.x + v0.y*v0.y + v0.z*v0.z + v0.w*v0.w +
               v1.x*v1.x + v1.y*v1.y + v1.z*v1.z + v1.w*v1.w;
#pragma unroll
    for (int msk = 1; msk < 32; msk <<= 1) {
        s  += __shfl_xor(s,  msk);
        sq += __shfl_xor(sq, msk);
    }
    const float mean = s * (1.f / D);
    const float var  = sq * (1.f / D) - mean * mean;
    const float rs   = __frsqrt_rn(var + 1e-5f);

    const float* gp = g + (size_t)t * D + lane * 8;
    const float* bp = b + (size_t)t * D + lane * 8;
    float* op = out + (size_t)row * D + lane * 8;
    float yy[8] = {v0.x, v0.y, v0.z, v0.w, v1.x, v1.y, v1.z, v1.w};
#pragma unroll
    for (int i = 0; i < 8; ++i)
        op[i] = (yy[i] - mean) * rs * gp[i] + bp[i];
}

// ------------------------------------------------------------------ launch --
extern "C" void kernel_launch(void* const* d_in, const int* in_sizes, int n_in,
                              void* d_out, int out_size, void* d_ws, size_t ws_size,
                              hipStream_t stream) {
    (void)in_sizes; (void)n_in; (void)out_size; (void)ws_size;

    const float* x    = (const float*)d_in[0];
    const float* Wq   = (const float*)d_in[1];
    const float* bq   = (const float*)d_in[2];
    const float* Wk   = (const float*)d_in[3];
    const float* bk   = (const float*)d_in[4];
    const float* Wv   = (const float*)d_in[5];
    const float* bv   = (const float*)d_in[6];
    const float* Wo   = (const float*)d_in[7];
    const float* bo   = (const float*)d_in[8];
    const float* ln_g = (const float*)d_in[9];
    const float* ln_b = (const float*)d_in[10];
    float* out = (float*)d_out;

    const size_t XND = (size_t)T * NN * D;   // 1,048,576 elems
    const size_t WDD = (size_t)T * D * D;    //   131,072 elems

    char* ws = (char*)d_ws;
    size_t off = 0;
    auto carve = [&](size_t bytes) { size_t o = off; off = (off + bytes + 255) & ~(size_t)255; return o; };
    __bf16* xb  = (__bf16*)(ws + carve(XND * 2));
    __bf16* wqb = (__bf16*)(ws + carve(WDD * 2));
    __bf16* wkb = (__bf16*)(ws + carve(WDD * 2));
    __bf16* wvb = (__bf16*)(ws + carve(WDD * 2));
    __bf16* wob = (__bf16*)(ws + carve(WDD * 2));
    __bf16* Qb  = (__bf16*)(ws + carve(XND * 2));
    __bf16* Kb  = (__bf16*)(ws + carve(XND * 2));
    __bf16* Vb  = (__bf16*)(ws + carve(XND * 2));
    __bf16* atb = (__bf16*)(ws + carve(XND * 2));
    float*  yb  = (float*)(ws + carve(XND * 4));

    convert_kernel<<<dim3((unsigned)(XND / 256)), 256, 0, stream>>>(x,  xb,  (int)XND);
    convert_kernel<<<dim3((unsigned)(WDD / 256)), 256, 0, stream>>>(Wq, wqb, (int)WDD);
    convert_kernel<<<dim3((unsigned)(WDD / 256)), 256, 0, stream>>>(Wk, wkb, (int)WDD);
    convert_kernel<<<dim3((unsigned)(WDD / 256)), 256, 0, stream>>>(Wv, wvb, (int)WDD);
    convert_kernel<<<dim3((unsigned)(WDD / 256)), 256, 0, stream>>>(Wo, wob, (int)WDD);

    proj_kernel<<<dim3(16, 4, T * 3), 256, 0, stream>>>(xb, wqb, wkb, wvb,
                                                        bq, bk, bv, Qb, Kb, Vb);
    attn_kernel<<<dim3(16, H, T), 256, 0, stream>>>(Qb, Kb, Vb, atb);
    oproj_kernel<<<dim3(16, 4, T), 256, 0, stream>>>(atb, wob, bo, x, yb);
    ln_kernel<<<dim3((T * NN) / 8), 256, 0, stream>>>(yb, ln_g, ln_b, out);
}
